// QLSTMEnhanced_65481071406488
// MI455X (gfx1250) — compile-verified, matrix operand order checked by reading, not run on hardware
//
#include <hip/hip_runtime.h>
#include <hip/hip_bf16.h>

typedef _Float16 v16h __attribute__((ext_vector_type(16)));
typedef float    v8f  __attribute__((ext_vector_type(8)));

#define T_STEPS 256
#define BATCH   512
#define HID     256
#define DIMX    256   // input dim (x part of combined)

// ---------------------------------------------------------------------------
// Kernel 0: wH[gn] = sum_{d=0..255} Wlin[gn][256+d]  (hidden-column weight sum;
// valid because all HIDDEN columns of hx carry the same scalar per row)
// ---------------------------------------------------------------------------
__global__ void qlstm_wsum(const float* __restrict__ Wlin, float* __restrict__ wH) {
    int lane = threadIdx.x;
    if (lane < 16) {
        const float* w = Wlin + (size_t)lane * 512 + 256;
        float s = 0.0f;
        #pragma unroll 8
        for (int d = 0; d < 256; ++d) s += w[d];
        wH[lane] = s;
    }
}

// ---------------------------------------------------------------------------
// Kernel 1: angX[(t*B+b)][gn] = inputs[t,b,:] . Wlin[gn,0:256] + blin[gn]+theta[gn]
// WMMA f32<=f16 with hi/lo split: A*B ~= Ah*Bh + Ah*Bl + Al*Bh (~fp32 accuracy).
// One 16x16 output tile per wave, 8 waves (128 rows) per block, K=256 in 8 chunks.
// ---------------------------------------------------------------------------
__global__ __launch_bounds__(256)
void qlstm_angx_gemm(const float* __restrict__ inputs,
                     const float* __restrict__ Wlin,
                     const float* __restrict__ blin,
                     const float* __restrict__ theta,
                     float* __restrict__ angX) {
    const int lane   = threadIdx.x & 31;
    const int wave   = threadIdx.x >> 5;
    const int row0   = (blockIdx.x * 8 + wave) * 16;   // first M row of this tile
    const int mrow   = row0 + (lane & 15);             // A row carried by this lane
    const int gn     = lane & 15;                      // B column carried by this lane
    const int hibase = (lane >> 4) << 3;               // K sub-offset for lane-halves

    const float* __restrict__ arow = inputs + (size_t)mrow * DIMX;
    const float* __restrict__ brow = Wlin   + (size_t)gn   * 512;   // x-weights: d<256

    v8f acc = {};
    #pragma unroll
    for (int c = 0; c < 8; ++c) {
        const int kc = c * 32;
        v16h ahi, alo, bhi, blo;
        // ISA 16-bit A/B layout: element e -> K = kc + (e&8)*2 + (e&7) + hibase
        // => two contiguous 8-float runs per lane (float4-friendly).
        #pragma unroll
        for (int e = 0; e < 16; ++e) {
            const int K = kc + ((e & 8) << 1) + (e & 7) + hibase;
            float av = arow[K];
            _Float16 ah = (_Float16)av;
            ahi[e] = ah;
            alo[e] = (_Float16)(av - (float)ah);
            float bv = brow[K];
            _Float16 bh = (_Float16)bv;
            bhi[e] = bh;
            blo[e] = (_Float16)(bv - (float)bh);
        }
        acc = __builtin_amdgcn_wmma_f32_16x16x32_f16(false, ahi, false, bhi, (short)0, acc, false, false);
        acc = __builtin_amdgcn_wmma_f32_16x16x32_f16(false, ahi, false, blo, (short)0, acc, false, false);
        acc = __builtin_amdgcn_wmma_f32_16x16x32_f16(false, alo, false, bhi, (short)0, acc, false, false);
    }

    const float bias = blin[gn] + theta[gn];
    #pragma unroll
    for (int r = 0; r < 8; ++r) {
        const int M = row0 + r + hibase;   // C/D layout: VGPR r, lane half selects M+8
        angX[(size_t)M * 16 + gn] = acc[r] + bias;
    }
}

// ---------------------------------------------------------------------------
// Kernel 2: scalar recurrence over T. Quantum circuit reduced analytically:
//   z0=C1C2C3, z1=C0C1, z2=C0C1C2, z3=C0C1C2C3  with Cw = cos(angle_w)
// (CNOT-ring permutes basis states; each output bit is a parity of independent
//  RX bits, so <Z_n> = prod over that parity set of cos(theta_w).)
// One lane per batch row; 16 blocks x 32 lanes so each wave owns a SIMD.
// ---------------------------------------------------------------------------
__device__ __forceinline__ float sigmoidf_(float x) {
    return 1.0f / (1.0f + __expf(-x));
}

__global__ void qlstm_recur(const float* __restrict__ angX,
                            const float* __restrict__ wH,
                            const float* __restrict__ rW,
                            const float* __restrict__ rb,
                            float* __restrict__ hseq,
                            float* __restrict__ cfin) {
    const int b = blockIdx.x * blockDim.x + threadIdx.x;   // 0..511 exact
    float wh[16], rw[16], rbv[4];
    #pragma unroll
    for (int j = 0; j < 16; ++j) { wh[j] = wH[j]; rw[j] = rW[j]; }
    #pragma unroll
    for (int g = 0; g < 4; ++g) rbv[g] = rb[g];

    float h = 0.0f, c = 0.0f;
    for (int t = 0; t < T_STEPS; ++t) {
        const float* __restrict__ a = angX + ((size_t)t * BATCH + b) * 16;
        float q[4];
        #pragma unroll
        for (int g = 0; g < 4; ++g) {
            const int k = g * 4;
            float C0 = __cosf(a[k + 0] + h * wh[k + 0]);
            float C1 = __cosf(a[k + 1] + h * wh[k + 1]);
            float C2 = __cosf(a[k + 2] + h * wh[k + 2]);
            float C3 = __cosf(a[k + 3] + h * wh[k + 3]);
            float z1 = C0 * C1;
            float z2 = z1 * C2;
            float z3 = z2 * C3;
            float z0 = C1 * C2 * C3;
            float s  = z0 * rw[k + 0] + z1 * rw[k + 1] + z2 * rw[k + 2] + z3 * rw[k + 3] + rbv[g];
            q[g] = sigmoidf_(s);
        }
        // reference double-applies the nonlinearity on q columns
        float f  = sigmoidf_(q[0]);
        float i  = sigmoidf_(q[1]);
        float gg = tanhf(q[2]);
        float o  = sigmoidf_(q[3]);
        c = f * c + i * gg;
        h = o * tanhf(c);
        hseq[(size_t)t * BATCH + b] = h;
    }
    cfin[b] = c;
}

// ---------------------------------------------------------------------------
// Kernel 3: broadcast scalars into (T,B,256) ys + (B,256) hx + (B,256) cx.
// float4 stores; 64 consecutive lanes share one hseq value (L2-broadcast read).
// This kernel moves the 134 MB that sets the runtime floor.
// ---------------------------------------------------------------------------
__global__ __launch_bounds__(256)
void qlstm_broadcast(const float* __restrict__ hseq,
                     const float* __restrict__ cfin,
                     float4* __restrict__ out) {
    const int TBH4 = T_STEPS * BATCH * (HID / 4);   // 8,388,608
    const int BH4  = BATCH * (HID / 4);             // 32,768
    const int idx  = blockIdx.x * blockDim.x + threadIdx.x;
    const int total4 = TBH4 + 2 * BH4;
    if (idx >= total4) return;

    float v;
    if (idx < TBH4) {
        int t = idx / (BATCH * (HID / 4));
        int b = (idx / (HID / 4)) % BATCH;
        v = hseq[(size_t)t * BATCH + b];
    } else if (idx < TBH4 + BH4) {
        int b = (idx - TBH4) / (HID / 4);
        v = hseq[(size_t)(T_STEPS - 1) * BATCH + b];
    } else {
        int b = (idx - TBH4 - BH4) / (HID / 4);
        v = cfin[b];
    }
    out[idx] = make_float4(v, v, v, v);
}

// ---------------------------------------------------------------------------
extern "C" void kernel_launch(void* const* d_in, const int* in_sizes, int n_in,
                              void* d_out, int out_size, void* d_ws, size_t ws_size,
                              hipStream_t stream) {
    const float* inputs = (const float*)d_in[0];   // (256,512,256)
    const float* Wlin   = (const float*)d_in[1];   // (4,4,512)
    const float* blin   = (const float*)d_in[2];   // (4,4)
    const float* theta  = (const float*)d_in[3];   // (4,4)
    const float* rW     = (const float*)d_in[4];   // (4,4)
    const float* rb     = (const float*)d_in[5];   // (4,)

    char* ws = (char*)d_ws;
    float* angX = (float*)ws;                                   // T*B*16 f32 = 8 MB
    float* hseq = (float*)(ws + (size_t)8 * 1024 * 1024 + 128 * 1024);  // T*B f32
    float* cfin = (float*)(ws + (size_t)9 * 1024 * 1024);               // B f32
    float* wH   = (float*)(ws + (size_t)9 * 1024 * 1024 + 4096);        // 16 f32

    qlstm_wsum<<<1, 32, 0, stream>>>(Wlin, wH);

    // M = T*B = 131072 rows; 128 rows per block (8 waves x 16-row WMMA tile)
    qlstm_angx_gemm<<<(T_STEPS * BATCH) / 128, 256, 0, stream>>>(
        inputs, Wlin, blin, theta, angX);

    qlstm_recur<<<BATCH / 32, 32, 0, stream>>>(angX, wH, rW, rb, hseq, cfin);

    const int total4 = T_STEPS * BATCH * (HID / 4) + 2 * BATCH * (HID / 4);
    qlstm_broadcast<<<(total4 + 255) / 256, 256, 0, stream>>>(
        hseq, cfin, (float4*)d_out);
}
